// RMTAttention_56513179680994
// MI455X (gfx1250) — matrix operand order repaired and checked by reading, not compiled, not run on hardware
//
#include <hip/hip_runtime.h>
#include <hip/hip_bf16.h>
#include <math.h>

// ---------------- problem constants ----------------
#define Bc     2
#define Sc     2048
#define Dc     1024
#define Hc     16
#define HKc    4
#define HDc    64
#define DKc    64
#define DVc    64
#define CHUNKc 64
#define NCHUNK (Sc / CHUNKc)
#define ROPE_BASE 500000.0f
#define MAX_LR    0.2f
#define MIN_DECAY 0.5f
#define MAX_NORM  30.0f
#define EPSc      1e-6f

typedef __bf16 bf16_t;
typedef bf16_t v16bf __attribute__((ext_vector_type(16)));
typedef bf16_t v8bf  __attribute__((ext_vector_type(8)));
typedef float  v8f   __attribute__((ext_vector_type(8)));

// ---------------- WMMA helpers ----------------
// A-fragment (16x32 bf16, MxK): lane L: m = L%16, half = L/16.
//   v[0..7]  = A[m][k0 + 8*half + 0..7]
//   v[8..15] = A[m][k0 + 16 + 8*half + 0..7]
__device__ __forceinline__ v16bf load_frag_a(const bf16_t* row, int k0, int half) {
  v8bf lo = *reinterpret_cast<const v8bf*>(row + k0 + half * 8);
  v8bf hi = *reinterpret_cast<const v8bf*>(row + k0 + 16 + half * 8);
  v16bf a;
#pragma unroll
  for (int i = 0; i < 8; ++i) { a[i] = lo[i]; a[i + 8] = hi[i]; }
  return a;
}

// B-fragment (32x16 bf16, KxN), fed from Bt[N][K] (i.e. column n is a row of Bt):
// lane L: n = L%16, half = L/16;  v[0..7] = B[k0 + 8*half + 0..7][n], v[8..15] = B[k0 + 16 + 8*half + 0..7][n]
__device__ __forceinline__ v16bf load_frag_b(const bf16_t* col, int k0, int half) {
  v8bf lo = *reinterpret_cast<const v8bf*>(col + k0 + half * 8);
  v8bf hi = *reinterpret_cast<const v8bf*>(col + k0 + 16 + half * 8);
  v16bf b;
#pragma unroll
  for (int i = 0; i < 8; ++i) { b[i] = lo[i]; b[i + 8] = hi[i]; }
  return b;
}

__device__ __forceinline__ v8f wmma_bf16(v16bf a, v16bf b, v8f c) {
  return __builtin_amdgcn_wmma_f32_16x16x32_bf16(false, a, false, b, (short)0, c,
                                                 false, false);
}

__device__ __forceinline__ float sigmoidf_(float x) { return 1.0f / (1.0f + __expf(-x)); }

// ---------------- conversion kernels ----------------
__global__ void cvt_bf16_kernel(const float* __restrict__ in, bf16_t* __restrict__ out, int n) {
  int i = blockIdx.x * 256 + threadIdx.x;
  if (i < n) out[i] = (bf16_t)in[i];
}

// in[K][N] -> out[N][K] (bf16)
__global__ void cvt_transpose_kernel(const float* __restrict__ in, bf16_t* __restrict__ out,
                                     int K, int N) {
  int i = blockIdx.x * 256 + threadIdx.x;
  if (i < K * N) {
    int k = i / N, n = i % N;
    out[(size_t)n * K + k] = (bf16_t)in[i];
  }
}

// RoPE: in fp32 [B,S,nH,HD] -> out bf16, rotated
__global__ void rope_kernel(const float* __restrict__ in, bf16_t* __restrict__ out, int nH) {
  int idx = blockIdx.x * 256 + threadIdx.x;   // over B*S*nH*(HD/2)
  int i = idx & 31;
  int rest = idx >> 5;                        // (b*S + s)*nH + hh
  if (rest >= Bc * Sc * nH) return;
  int s = (rest / nH) % Sc;
  size_t base = (size_t)rest * HDc;
  float inv = __expf(-(2.0f * (float)i / (float)HDc) * __logf(ROPE_BASE));
  float th = (float)s * inv;
  float cs = cosf(th), sn = sinf(th);
  float u0 = in[base + i], u1 = in[base + i + 32];
  out[base + i]      = (bf16_t)(u0 * cs - u1 * sn);
  out[base + i + 32] = (bf16_t)(u1 * cs + u0 * sn);
}

// v fp32 [B,S,HK,HD] -> vT bf16 [B,HK,HD,S]
__global__ void vtrans_kernel(const float* __restrict__ vf, bf16_t* __restrict__ vT) {
  int idx = blockIdx.x * 256 + threadIdx.x;   // B*S*HK*HD
  if (idx >= Bc * Sc * HKc * HDc) return;
  int d  = idx % HDc;
  int kh = (idx / HDc) % HKc;
  int s  = (idx / (HDc * HKc)) % Sc;
  int b  = idx / (HDc * HKc * Sc);
  vT[(((size_t)b * HKc + kh) * HDc + d) * Sc + s] = (bf16_t)vf[idx];
}

// ---------------- WMMA GEMM: C[M,N] (+)= A[M,K] * Bt[N,K]^T ----------------
// block = 256 threads = 8 waves; wave w owns a 16x64 C strip (4 accumulators),
// reusing its A fragment across the 4 N-subtiles (~102 FLOP/B from L2).
// Ping/pong software pipeline (K must be a multiple of 64): each stage's
// fragments are written directly by loads (no register rotation), and the
// next stage's loads are issued before the current stage's WMMAs.
// grid: (N/64, M/128)
__global__ __launch_bounds__(256) void gemm_bf16_kernel(const bf16_t* __restrict__ A,
                                                        const bf16_t* __restrict__ Bt,
                                                        float* __restrict__ C,
                                                        int M, int N, int K, int accum) {
  int tid = threadIdx.x;
  int w = tid >> 5, lane = tid & 31;
  int l16 = lane & 15, half = lane >> 4;
  int row0 = blockIdx.y * 128 + w * 16;
  int col0 = blockIdx.x * 64;

  const bf16_t* arow = A + (size_t)(row0 + l16) * K;
  const bf16_t* brow0 = Bt + (size_t)(col0 + 0 * 16 + l16) * K;
  const bf16_t* brow1 = Bt + (size_t)(col0 + 1 * 16 + l16) * K;
  const bf16_t* brow2 = Bt + (size_t)(col0 + 2 * 16 + l16) * K;
  const bf16_t* brow3 = Bt + (size_t)(col0 + 3 * 16 + l16) * K;

  v8f acc0 = {}, acc1 = {}, acc2 = {}, acc3 = {};
  if (accum) {
#pragma unroll
    for (int r = 0; r < 8; ++r) {
      size_t ro = (size_t)(row0 + r + 8 * half) * N + col0 + l16;
      acc0[r] = C[ro];
      acc1[r] = C[ro + 16];
      acc2[r] = C[ro + 32];
      acc3[r] = C[ro + 48];
    }
  }

  // even-stage fragments for k0 = 0
  v16bf ea  = load_frag_a(arow, 0, half);
  v16bf eb0 = load_frag_b(brow0, 0, half);
  v16bf eb1 = load_frag_b(brow1, 0, half);
  v16bf eb2 = load_frag_b(brow2, 0, half);
  v16bf eb3 = load_frag_b(brow3, 0, half);

  for (int k0 = 0; k0 < K; k0 += 64) {
    if (k0 + 192 < K) __builtin_prefetch(arow + k0 + 192, 0, 1);
    // odd-stage loads (k0+32) issue before even-stage WMMAs
    int ko = k0 + 32;
    v16bf oa  = load_frag_a(arow, ko, half);
    v16bf ob0 = load_frag_b(brow0, ko, half);
    v16bf ob1 = load_frag_b(brow1, ko, half);
    v16bf ob2 = load_frag_b(brow2, ko, half);
    v16bf ob3 = load_frag_b(brow3, ko, half);
    acc0 = wmma_bf16(ea, eb0, acc0);
    acc1 = wmma_bf16(ea, eb1, acc1);
    acc2 = wmma_bf16(ea, eb2, acc2);
    acc3 = wmma_bf16(ea, eb3, acc3);
    // next even-stage loads (k0+64) issue before odd-stage WMMAs
    int ke = k0 + 64;
    if (ke < K) {
      ea  = load_frag_a(arow, ke, half);
      eb0 = load_frag_b(brow0, ke, half);
      eb1 = load_frag_b(brow1, ke, half);
      eb2 = load_frag_b(brow2, ke, half);
      eb3 = load_frag_b(brow3, ke, half);
    }
    acc0 = wmma_bf16(oa, ob0, acc0);
    acc1 = wmma_bf16(oa, ob1, acc1);
    acc2 = wmma_bf16(oa, ob2, acc2);
    acc3 = wmma_bf16(oa, ob3, acc3);
  }
#pragma unroll
  for (int r = 0; r < 8; ++r) {
    size_t ro = (size_t)(row0 + r + 8 * half) * N + col0 + l16;
    C[ro]      = acc0[r];
    C[ro + 16] = acc1[r];
    C[ro + 32] = acc2[r];
    C[ro + 48] = acc3[r];
  }
}

// ---------------- flash attention (causal, GQA, bf16 WMMA) ----------------
// grid: (B*H, S/128); block 256 = 8 waves; wave w owns q rows [qRow0, qRow0+16)
__global__ __launch_bounds__(256) void attn_kernel(const bf16_t* __restrict__ qb,
                                                   const bf16_t* __restrict__ kb,
                                                   const bf16_t* __restrict__ vTb,
                                                   bf16_t* __restrict__ attn_b) {
  __shared__ __align__(16) bf16_t Pb[8][16][64];

  int tid = threadIdx.x;
  int w = tid >> 5, lane = tid & 31;
  int l16 = lane & 15, half = lane >> 4;
  int bh = blockIdx.x;
  int b = bh / Hc, h = bh % Hc;
  int kh = h / (Hc / HKc);
  int qRow0 = blockIdx.y * 128 + w * 16;

  const bf16_t* qrow = qb + ((size_t)(b * Sc + qRow0 + l16) * Hc + h) * HDc;
  v16bf aq0 = load_frag_a(qrow, 0, half);
  v16bf aq1 = load_frag_a(qrow, 32, half);

  v8f acc[4];
#pragma unroll
  for (int ns = 0; ns < 4; ++ns) acc[ns] = (v8f){};
  float mrow[8], lrow[8];
#pragma unroll
  for (int r = 0; r < 8; ++r) { mrow[r] = -1e30f; lrow[r] = 0.0f; }

  const bf16_t* vbase = vTb + ((size_t)(b * HKc + kh) * HDc) * Sc;
  int ktEnd = qRow0 + 15;

  for (int kt = 0; kt <= ktEnd; kt += 64) {
    // ---- scores S = Q K^T / 8 ----
    v8f s[4];
#pragma unroll
    for (int ns = 0; ns < 4; ++ns) {
      int key = kt + ns * 16 + l16;
      const bf16_t* krow = kb + ((size_t)(b * Sc + key) * HKc + kh) * HDc;
      v16bf b0 = load_frag_b(krow, 0, half);
      v16bf b1 = load_frag_b(krow, 32, half);
      v8f c = {};
      c = wmma_bf16(aq0, b0, c);
      c = wmma_bf16(aq1, b1, c);
      s[ns] = c;
    }
    // scale + causal mask
#pragma unroll
    for (int ns = 0; ns < 4; ++ns) {
      int key = kt + ns * 16 + l16;
#pragma unroll
      for (int r = 0; r < 8; ++r) {
        int q = qRow0 + r + 8 * half;
        float v = s[ns][r] * 0.125f;
        s[ns][r] = (key <= q) ? v : -1e30f;
      }
    }
    // ---- online softmax (row reductions across 16 lanes of each half) ----
#pragma unroll
    for (int r = 0; r < 8; ++r) {
      float rm = fmaxf(fmaxf(s[0][r], s[1][r]), fmaxf(s[2][r], s[3][r]));
#pragma unroll
      for (int off = 1; off < 16; off <<= 1) rm = fmaxf(rm, __shfl_xor(rm, off, 32));
      float nm = fmaxf(mrow[r], rm);
      float scl = __expf(mrow[r] - nm);
      mrow[r] = nm;
      float rs = 0.0f;
#pragma unroll
      for (int ns = 0; ns < 4; ++ns) {
        float p = __expf(s[ns][r] - nm);
        s[ns][r] = p;
        rs += p;
      }
#pragma unroll
      for (int off = 1; off < 16; off <<= 1) rs += __shfl_xor(rs, off, 32);
      lrow[r] = lrow[r] * scl + rs;
#pragma unroll
      for (int ns = 0; ns < 4; ++ns) acc[ns][r] *= scl;
    }
    // ---- P to LDS (bf16) then reload in A-fragment layout (intra-wave) ----
#pragma unroll
    for (int ns = 0; ns < 4; ++ns)
#pragma unroll
      for (int r = 0; r < 8; ++r)
        Pb[w][r + 8 * half][ns * 16 + l16] = (bf16_t)s[ns][r];
    asm volatile("s_wait_dscnt 0" ::: "memory");
    const bf16_t* prow = &Pb[w][l16][0];
    v16bf pa0 = load_frag_a(prow, 0, half);
    v16bf pa1 = load_frag_a(prow, 32, half);
    asm volatile("" ::: "memory");
    // ---- O += P V ----
#pragma unroll
    for (int ns = 0; ns < 4; ++ns) {
      const bf16_t* vrow = vbase + (size_t)(ns * 16 + l16) * Sc + kt;
      v16bf vb0 = load_frag_b(vrow, 0, half);
      v16bf vb1 = load_frag_b(vrow, 32, half);
      acc[ns] = wmma_bf16(pa0, vb0, acc[ns]);
      acc[ns] = wmma_bf16(pa1, vb1, acc[ns]);
    }
  }
  // ---- epilogue: normalize and store bf16 [B,S,H*HD] ----
#pragma unroll
  for (int r = 0; r < 8; ++r) {
    float inv = 1.0f / fmaxf(lrow[r], 1e-30f);
    size_t rowoff = ((size_t)(b * Sc + qRow0 + r + 8 * half) * Hc + h) * HDc;
#pragma unroll
    for (int ns = 0; ns < 4; ++ns)
      attn_b[rowoff + ns * 16 + l16] = (bf16_t)(acc[ns][r] * inv);
  }
}

// ---------------- sequential chunked delta-rule memory scan ----------------
// grid: (B); block 256. M lives in LDS; 32 sequential chunks.
__global__ __launch_bounds__(256) void memscan_kernel(
    const float* __restrict__ z, const float* __restrict__ M_init,
    const float* __restrict__ w_eta, const float* __restrict__ b_eta,
    const float* __restrict__ w_alpha, const float* __restrict__ b_alpha,
    const float* __restrict__ w_gate, const float* __restrict__ b_gate,
    bf16_t* __restrict__ memout_b) {
  __shared__ float Ms[DVc * DKc];
  __shared__ float Ch[CHUNKc * DKc];
  __shared__ float Out[CHUNKc * DVc];
  __shared__ float tokE[64], tokA[64], tokG[64], tokN[64];
  __shared__ float kmean[DKc], vt[DVc], Mk[DVc];
  __shared__ float red[256];
  __shared__ float sc[3];

  int b = blockIdx.x;
  int tid = threadIdx.x;
  for (int i = tid; i < DVc * DKc; i += 256) Ms[i] = M_init[i];
  __syncthreads();
  float be = b_eta[0], ba = b_alpha[0], bg = b_gate[0];

  for (int c = 0; c < NCHUNK; ++c) {
    const float* zc = z + ((size_t)b * Sc + c * CHUNKc) * DKc;
    for (int i = tid; i < CHUNKc * DKc; i += 256) Ch[i] = zc[i];
    __syncthreads();

    if (tid < 64) {
      int t = tid;
      float de = 0, da = 0, dg = 0, nn = 0;
      for (int d = 0; d < DKc; ++d) {
        float x = Ch[t * DKc + d];
        de += x * w_eta[d]; da += x * w_alpha[d]; dg += x * w_gate[d]; nn += x * x;
      }
      tokE[t] = sigmoidf_(de + be) * MAX_LR;
      tokA[t] = MIN_DECAY + sigmoidf_(da + ba) * (1.0f - MIN_DECAY);
      tokG[t] = sigmoidf_(dg + bg);
      tokN[t] = fmaxf(sqrtf(nn), EPSc);
    }
    __syncthreads();

    // retrieve: Out[t][v] = ch[t] . M[v]
    {
      int t = tid >> 2, vb0 = (tid & 3) * 16;
      for (int j = 0; j < 16; ++j) {
        int v = vb0 + j;
        float s = 0;
        for (int d = 0; d < DKc; ++d) s += Ch[t * DKc + d] * Ms[v * DKc + d];
        Out[t * DVc + v] = s;
        memout_b[((size_t)b * Sc + c * CHUNKc + t) * DVc + v] = (bf16_t)s;
      }
    }
    __syncthreads();

    if (tid < 64) {
      int d = tid; float s = 0;
      for (int t = 0; t < 64; ++t) s += Ch[t * DKc + d] / tokN[t];
      kmean[d] = s * (1.0f / 64.0f);
    } else if (tid < 128) {
      int v = tid - 64; float s = 0;
      for (int t = 0; t < 64; ++t) s += Out[t * DVc + v];
      vt[v] = s * (1.0f / 64.0f);
    } else if (tid == 128) {
      float s = 0; for (int t = 0; t < 64; ++t) s += tokE[t]; sc[0] = s * (1.0f / 64.0f);
    } else if (tid == 129) {
      float s = 0; for (int t = 0; t < 64; ++t) s += tokA[t]; sc[1] = s * (1.0f / 64.0f);
    } else if (tid == 130) {
      float s = 0; for (int t = 0; t < 64; ++t) s += tokG[t]; sc[2] = s * (1.0f / 64.0f);
    }
    __syncthreads();

    if (tid < 64) {
      int v = tid; float s = 0;
      for (int d = 0; d < DKc; ++d) s += Ms[v * DKc + d] * kmean[d];
      Mk[v] = s;
    }
    __syncthreads();

    float eg = sc[0] * sc[2], al = sc[1];
    float part = 0;
    for (int i = tid; i < DVc * DKc; i += 256) {
      int v = i >> 6, d = i & 63;
      float upd = (vt[v] - Mk[v]) * kmean[d];
      float mn = al * Ms[i] + eg * upd;
      Ms[i] = mn;
      part += mn * mn;
    }
    red[tid] = part;
    __syncthreads();
    for (int s2 = 128; s2 > 0; s2 >>= 1) {
      if (tid < s2) red[tid] += red[tid + s2];
      __syncthreads();
    }
    float fro = sqrtf(red[0]);
    float scale = fminf(1.0f, MAX_NORM / fmaxf(fro, EPSc));
    for (int i = tid; i < DVc * DKc; i += 256) Ms[i] *= scale;
    __syncthreads();
  }
}

// ---------------- launch ----------------
extern "C" void kernel_launch(void* const* d_in, const int* in_sizes, int n_in,
                              void* d_out, int out_size, void* d_ws, size_t ws_size,
                              hipStream_t stream) {
  (void)in_sizes; (void)n_in; (void)out_size; (void)ws_size;
  const float* x        = (const float*)d_in[0];
  const float* Wq       = (const float*)d_in[1];
  const float* Wk       = (const float*)d_in[2];
  const float* Wv       = (const float*)d_in[3];
  const float* Wo       = (const float*)d_in[4];
  const float* W_memin  = (const float*)d_in[5];
  const float* W_memout = (const float*)d_in[6];
  const float* M_init   = (const float*)d_in[7];
  const float* w_eta    = (const float*)d_in[8];
  const float* b_eta    = (const float*)d_in[9];
  const float* w_alpha  = (const float*)d_in[10];
  const float* b_alpha  = (const float*)d_in[11];
  const float* w_gate   = (const float*)d_in[12];
  const float* b_gate   = (const float*)d_in[13];
  float* out = (float*)d_out;

  char* ws = (char*)d_ws;
  size_t off = 0;
  auto alloc = [&](size_t bytes) -> char* {
    char* p = ws + off;
    off = (off + bytes + 255) & ~(size_t)255;
    return p;
  };

  const size_t BSD = (size_t)Bc * Sc * Dc;          // 4M
  const size_t BSQ = (size_t)Bc * Sc * Hc * HDc;    // 4M
  const size_t BSK = (size_t)Bc * Sc * HKc * HDc;   // 1M
  const size_t BSM = (size_t)Bc * Sc * DKc;         // 256K

  bf16_t* xb     = (bf16_t*)alloc(BSD * 2);
  bf16_t* WqT    = (bf16_t*)alloc((size_t)Dc * Hc * HDc * 2);
  bf16_t* WkT    = (bf16_t*)alloc((size_t)Dc * HKc * HDc * 2);
  bf16_t* WvT    = (bf16_t*)alloc((size_t)Dc * HKc * HDc * 2);
  bf16_t* WoT    = (bf16_t*)alloc((size_t)Hc * HDc * Dc * 2);
  bf16_t* WminT  = (bf16_t*)alloc((size_t)Dc * DKc * 2);
  bf16_t* WmoutT = (bf16_t*)alloc((size_t)DVc * Dc * 2);
  float*  qf     = (float*)alloc(BSQ * 4);
  float*  kf     = (float*)alloc(BSK * 4);
  float*  vf     = (float*)alloc(BSK * 4);
  float*  zf     = (float*)alloc(BSM * 4);
  bf16_t* qb     = (bf16_t*)alloc(BSQ * 2);
  bf16_t* kb     = (bf16_t*)alloc(BSK * 2);
  bf16_t* vTb    = (bf16_t*)alloc(BSK * 2);
  bf16_t* attn_b = (bf16_t*)alloc(BSQ * 2);
  bf16_t* memo_b = (bf16_t*)alloc((size_t)Bc * Sc * DVc * 2);

  // 1) conversions
  cvt_bf16_kernel<<<(int)(BSD / 256), 256, 0, stream>>>(x, xb, (int)BSD);
  cvt_transpose_kernel<<<(Dc * Hc * HDc) / 256, 256, 0, stream>>>(Wq, WqT, Dc, Hc * HDc);
  cvt_transpose_kernel<<<(Dc * HKc * HDc) / 256, 256, 0, stream>>>(Wk, WkT, Dc, HKc * HDc);
  cvt_transpose_kernel<<<(Dc * HKc * HDc) / 256, 256, 0, stream>>>(Wv, WvT, Dc, HKc * HDc);
  cvt_transpose_kernel<<<(Hc * HDc * Dc) / 256, 256, 0, stream>>>(Wo, WoT, Hc * HDc, Dc);
  cvt_transpose_kernel<<<(Dc * DKc) / 256, 256, 0, stream>>>(W_memin, WminT, Dc, DKc);
  cvt_transpose_kernel<<<(DVc * Dc) / 256, 256, 0, stream>>>(W_memout, WmoutT, DVc, Dc);

  // 2) projections (M = B*S = 4096); grid (N/64, M/128)
  const int M = Bc * Sc;
  gemm_bf16_kernel<<<dim3((Hc * HDc) / 64, M / 128), 256, 0, stream>>>(xb, WqT, qf, M, Hc * HDc, Dc, 0);
  gemm_bf16_kernel<<<dim3((HKc * HDc) / 64, M / 128), 256, 0, stream>>>(xb, WkT, kf, M, HKc * HDc, Dc, 0);
  gemm_bf16_kernel<<<dim3((HKc * HDc) / 64, M / 128), 256, 0, stream>>>(xb, WvT, vf, M, HKc * HDc, Dc, 0);
  gemm_bf16_kernel<<<dim3(DKc / 64, M / 128), 256, 0, stream>>>(xb, WminT, zf, M, DKc, Dc, 0);

  // 3) rope + v transpose
  rope_kernel<<<(int)((BSQ / 2) / 256), 256, 0, stream>>>(qf, qb, Hc);
  rope_kernel<<<(int)((BSK / 2) / 256), 256, 0, stream>>>(kf, kb, HKc);
  vtrans_kernel<<<(int)(BSK / 256), 256, 0, stream>>>(vf, vTb);

  // 4) attention
  attn_kernel<<<dim3(Bc * Hc, Sc / 128), 256, 0, stream>>>(qb, kb, vTb, attn_b);

  // 5) sequential memory scan
  memscan_kernel<<<Bc, 256, 0, stream>>>(zf, M_init, w_eta, b_eta, w_alpha, b_alpha,
                                         w_gate, b_gate, memo_b);

  // 6) output GEMMs: out = attn @ Wo ; out += mem @ W_memout
  gemm_bf16_kernel<<<dim3(Dc / 64, M / 128), 256, 0, stream>>>(attn_b, WoT, out, M, Dc, Hc * HDc, 0);
  gemm_bf16_kernel<<<dim3(Dc / 64, M / 128), 256, 0, stream>>>(memo_b, WmoutT, out, M, Dc, DVc, 1);
}
